// MQALayer_14405320311598
// MI455X (gfx1250) — compile-verified, matrix operand order checked by reading, not run on hardware
//
#include <hip/hip_runtime.h>

#define T_TOK 2048
#define HIDN  2048
#define QLRK  1536
#define NHEAD 16
#define NDIM  128
#define HDIM  192
#define NGRP  4
#define OLRK  512
#define GDIM  768
#define EPSF  1e-6f
#define NEGI  -1e30f

typedef __bf16 bf16_t;
typedef __attribute__((ext_vector_type(16))) __bf16 v16bf;
typedef __attribute__((ext_vector_type(8)))  float  v8f;

union FragU {
  v16bf v;
  uint4 q[2];
  unsigned short h[16];
};

__device__ inline bf16_t f2bf(float f) {
  union { float f; unsigned u; } a; a.f = f;
  unsigned u = a.u + 0x7FFFu + ((a.u >> 16) & 1u);
  union { unsigned short s; bf16_t b; } r; r.s = (unsigned short)(u >> 16);
  return r.b;
}

__device__ inline v8f wmma_bf16(v16bf a, v16bf b, v8f c) {
  return __builtin_amdgcn_wmma_f32_16x16x32_bf16(false, a, false, b, (short)0, c, false, false);
}

// 16-bit A/B fragment layout (16x32 / 32x16): lane holds row/col = lane&15,
// element e -> K = (e>>3)*16 + ((lane>>4)<<3) + (e&7): two contiguous 16B runs.
__device__ inline v16bf load_frag(const bf16_t* __restrict__ base, int ld,
                                  int rc0, int k0, int lane) {
  const bf16_t* p = base + (size_t)(rc0 + (lane & 15)) * ld + k0 + ((lane >> 4) << 3);
  FragU f;
  f.q[0] = *(const uint4*)(p);
  f.q[1] = *(const uint4*)(p + 16);
  return f.v;
}

__global__ void cvt_bf16(const float* __restrict__ in, bf16_t* __restrict__ out, int n) {
  int i = blockIdx.x * blockDim.x + threadIdx.x;
  if (i < n) out[i] = f2bf(in[i]);
}

// C[m,n] = sum_k A[m,k]*B[n,k]  (A:[M,K] row-major, B:[N,K] row-major => A @ B^T)
// One wave computes a 64x64 tile. OBF16: compile-time output type selection.
template <bool OBF16>
__global__ __launch_bounds__(32) void gemm_bf16(
    const bf16_t* __restrict__ A, int lda,
    const bf16_t* __restrict__ B, int ldb,
    void* __restrict__ Cout, int ldc, int K) {
  const int lane = threadIdx.x & 31;
  const int n0 = blockIdx.x * 64;
  const int m0 = blockIdx.y * 64;
  v8f acc[4][4];
#pragma unroll
  for (int i = 0; i < 4; ++i)
#pragma unroll
    for (int j = 0; j < 4; ++j)
#pragma unroll
      for (int r = 0; r < 8; ++r) acc[i][j][r] = 0.f;
  for (int k0 = 0; k0 < K; k0 += 32) {
    v16bf a[4], b[4];
#pragma unroll
    for (int i = 0; i < 4; ++i) a[i] = load_frag(A, lda, m0 + i * 16, k0, lane);
#pragma unroll
    for (int j = 0; j < 4; ++j) b[j] = load_frag(B, ldb, n0 + j * 16, k0, lane);
#pragma unroll
    for (int i = 0; i < 4; ++i)
#pragma unroll
      for (int j = 0; j < 4; ++j)
        acc[i][j] = wmma_bf16(a[i], b[j], acc[i][j]);
  }
  const int nn = lane & 15;
  const int rb = (lane >> 4) << 3;
#pragma unroll
  for (int i = 0; i < 4; ++i)
#pragma unroll
    for (int j = 0; j < 4; ++j)
#pragma unroll
      for (int r = 0; r < 8; ++r) {
        size_t idx = (size_t)(m0 + i * 16 + rb + r) * ldc + (n0 + j * 16 + nn);
        float v = acc[i][j][r];
        if (OBF16) ((bf16_t*)Cout)[idx] = f2bf(v);
        else       ((float*)Cout)[idx] = v;
      }
}

__global__ __launch_bounds__(256) void rmsnorm_rows(
    const float* __restrict__ in, const float* __restrict__ w,
    bf16_t* __restrict__ out, int C) {
  __shared__ float red[256];
  int row = blockIdx.x;
  const float* x = in + (size_t)row * C;
  float ss = 0.f;
  for (int i = threadIdx.x; i < C; i += 256) { float v = x[i]; ss += v * v; }
  red[threadIdx.x] = ss;
  __syncthreads();
  for (int s = 128; s > 0; s >>= 1) {
    if ((int)threadIdx.x < s) red[threadIdx.x] += red[threadIdx.x + s];
    __syncthreads();
  }
  float inv = rsqrtf(red[0] / (float)C + EPSF);
  for (int i = threadIdx.x; i < C; i += 256)
    out[(size_t)row * C + i] = f2bf(x[i] * inv * w[i]);
}

// per-(t,h) weightless RMSNorm + RoPE on dims [128,192) + fold attn scale. 1 wave each.
__global__ __launch_bounds__(32) void q_norm_rope(
    const float* __restrict__ qf, const int* __restrict__ pos,
    bf16_t* __restrict__ qb) {
  int th = blockIdx.x;
  int t = th / NHEAD;
  int lane = threadIdx.x;
  const float* x = qf + (size_t)th * HDIM;
  float v[6]; float ss = 0.f;
#pragma unroll
  for (int e = 0; e < 6; ++e) { v[e] = x[lane * 6 + e]; ss += v[e] * v[e]; }
#pragma unroll
  for (int m = 16; m; m >>= 1) ss += __shfl_xor(ss, m, 32);
  float inv = rsqrtf(ss / (float)HDIM + EPSF);
#pragma unroll
  for (int e = 0; e < 6; ++e) v[e] *= inv;
  float p = (float)pos[t];
#pragma unroll
  for (int e = 0; e < 6; e += 2) {
    int gd = lane * 6 + e;
    if (gd >= NDIM) {
      int pr = (gd - NDIM) >> 1;
      float ang = p * __powf(10000.0f, -(float)pr * (1.0f / 32.0f));
      float c = __cosf(ang), s = __sinf(ang);
      float x1 = v[e], x2 = v[e + 1];
      v[e] = x1 * c - x2 * s;
      v[e + 1] = x1 * s + x2 * c;
    }
  }
  const float scale = 0.07216878364870323f; // 192^-0.5 folded into q
#pragma unroll
  for (int e = 0; e < 6; ++e)
    qb[(size_t)th * HDIM + lane * 6 + e] = f2bf(v[e] * scale);
}

// RMSNorm(w) + RoPE; writes kv row-major [T,192] and transposed kvT [192,T].
__global__ __launch_bounds__(32) void kv_norm_rope(
    const float* __restrict__ kvf, const float* __restrict__ w,
    const int* __restrict__ pos, bf16_t* __restrict__ kvb,
    bf16_t* __restrict__ kvT) {
  int t = blockIdx.x;
  int lane = threadIdx.x;
  const float* x = kvf + (size_t)t * HDIM;
  float v[6]; float ss = 0.f;
#pragma unroll
  for (int e = 0; e < 6; ++e) { v[e] = x[lane * 6 + e]; ss += v[e] * v[e]; }
#pragma unroll
  for (int m = 16; m; m >>= 1) ss += __shfl_xor(ss, m, 32);
  float inv = rsqrtf(ss / (float)HDIM + EPSF);
#pragma unroll
  for (int e = 0; e < 6; ++e) v[e] = v[e] * inv * w[lane * 6 + e];
  float p = (float)pos[t];
#pragma unroll
  for (int e = 0; e < 6; e += 2) {
    int gd = lane * 6 + e;
    if (gd >= NDIM) {
      int pr = (gd - NDIM) >> 1;
      float ang = p * __powf(10000.0f, -(float)pr * (1.0f / 32.0f));
      float c = __cosf(ang), s = __sinf(ang);
      float x1 = v[e], x2 = v[e + 1];
      v[e] = x1 * c - x2 * s;
      v[e + 1] = x1 * s + x2 * c;
    }
  }
#pragma unroll
  for (int e = 0; e < 6; ++e) {
    bf16_t b = f2bf(v[e]);
    kvb[(size_t)t * HDIM + lane * 6 + e] = b;
    kvT[(size_t)(lane * 6 + e) * T_TOK + t] = b;
  }
}

// Flash-style causal MQA with sink. One wave per (head, 16-query-row tile).
// 32 keys/step; B-fragments staged in batches so loads clause and overlap WMMA.
__global__ __launch_bounds__(32) void attn_mqa(
    const bf16_t* __restrict__ qb, const bf16_t* __restrict__ kvb,
    const bf16_t* __restrict__ kvT, const float* __restrict__ sink,
    bf16_t* __restrict__ ob) {
  __shared__ __align__(16) bf16_t P[16 * 32];
  const int lane = threadIdx.x;
  const int tile = blockIdx.x;
  const int h = tile >> 7;        // / (T/16)
  const int it = tile & 127;
  const int m0 = it * 16;
  const int ldq = NHEAD * HDIM;   // 3072
  const int nn = lane & 15;
  const int kb8 = (lane >> 4) << 3;

  v16bf qa[6];
#pragma unroll
  for (int i = 0; i < 6; ++i) qa[i] = load_frag(qb + h * HDIM, ldq, m0, i * 32, lane);

  v8f O[12];
#pragma unroll
  for (int n = 0; n < 12; ++n)
#pragma unroll
    for (int r = 0; r < 8; ++r) O[n][r] = 0.f;
  float mrow[8], drow[8];
  float sk = sink[h];
#pragma unroll
  for (int r = 0; r < 8; ++r) { mrow[r] = sk; drow[r] = 1.f; } // sink term: exp(sk-sk)=1

  const int jlast = it >> 1;
  for (int j = 0; j <= jlast; ++j) {
    int s0 = j * 32;
    v8f c0, c1;
#pragma unroll
    for (int r = 0; r < 8; ++r) { c0[r] = 0.f; c1[r] = 0.f; }
    // QK: 4 batches of 3 fragments -> loads clause, drain with progressive waits
#pragma unroll
    for (int g = 0; g < 4; ++g) {
      const int srow = s0 + (g >= 2 ? 16 : 0);
      const int kk0 = (g & 1) * 3;
      v16bf bv[3];
#pragma unroll
      for (int q = 0; q < 3; ++q)
        bv[q] = load_frag(kvb, HDIM, srow, (kk0 + q) * 32, lane);
#pragma unroll
      for (int q = 0; q < 3; ++q) {
        if (g < 2) c0 = wmma_bf16(qa[kk0 + q], bv[q], c0);
        else       c1 = wmma_bf16(qa[kk0 + q], bv[q], c1);
      }
    }
    if (j == jlast) { // mask keys beyond the diagonal (covers odd-tile padding)
#pragma unroll
      for (int r = 0; r < 8; ++r) {
        int row = m0 + kb8 + r;
        if (s0 + nn > row)      c0[r] = NEGI;
        if (s0 + 16 + nn > row) c1[r] = NEGI;
      }
    }
    float alpha[8];
#pragma unroll
    for (int r = 0; r < 8; ++r) {
      float mx = fmaxf(c0[r], c1[r]);
      mx = fmaxf(mx, __shfl_xor(mx, 1, 32));
      mx = fmaxf(mx, __shfl_xor(mx, 2, 32));
      mx = fmaxf(mx, __shfl_xor(mx, 4, 32));
      mx = fmaxf(mx, __shfl_xor(mx, 8, 32));
      float nm = fmaxf(mrow[r], mx);
      alpha[r] = __expf(mrow[r] - nm);
      mrow[r] = nm;
      float p0 = __expf(c0[r] - nm);
      float p1 = __expf(c1[r] - nm);
      c0[r] = p0; c1[r] = p1;
      float sm = p0 + p1;
      sm += __shfl_xor(sm, 1, 32);
      sm += __shfl_xor(sm, 2, 32);
      sm += __shfl_xor(sm, 4, 32);
      sm += __shfl_xor(sm, 8, 32);
      drow[r] = drow[r] * alpha[r] + sm;
    }
#pragma unroll
    for (int n = 0; n < 12; ++n)
#pragma unroll
      for (int r = 0; r < 8; ++r) O[n][r] *= alpha[r];
    // stage P (16 rows x 32 keys) in LDS, re-read in A-fragment layout
#pragma unroll
    for (int r = 0; r < 8; ++r) {
      P[(kb8 + r) * 32 + nn]      = f2bf(c0[r]);
      P[(kb8 + r) * 32 + 16 + nn] = f2bf(c1[r]);
    }
    __syncthreads();
    FragU pf;
    pf.q[0] = *(const uint4*)(P + nn * 32 + kb8);
    pf.q[1] = *(const uint4*)(P + nn * 32 + kb8 + 16);
    v16bf pa = pf.v;
    // PV: 3 batches of 4 fragments from kvT (dim-major, [N,K] row-major)
#pragma unroll
    for (int g = 0; g < 3; ++g) {
      v16bf bv[4];
#pragma unroll
      for (int q = 0; q < 4; ++q)
        bv[q] = load_frag(kvT, T_TOK, (g * 4 + q) * 16, s0, lane);
#pragma unroll
      for (int q = 0; q < 4; ++q)
        O[g * 4 + q] = wmma_bf16(pa, bv[q], O[g * 4 + q]);
    }
    __syncthreads();
  }
#pragma unroll
  for (int n = 0; n < 12; ++n)
#pragma unroll
    for (int r = 0; r < 8; ++r) {
      size_t row = (size_t)(m0 + kb8 + r);
      size_t col = (size_t)h * HDIM + n * 16 + nn;
      ob[row * ldq + col] = f2bf(O[n][r] / drow[r]);
    }
}

extern "C" void kernel_launch(void* const* d_in, const int* in_sizes, int n_in,
                              void* d_out, int out_size, void* d_ws, size_t ws_size,
                              hipStream_t stream) {
  (void)in_sizes; (void)n_in; (void)out_size; (void)ws_size;
  const float* x    = (const float*)d_in[0];
  const int*   pos  = (const int*)d_in[1];
  const float* wq_a = (const float*)d_in[2];
  const float* qnw  = (const float*)d_in[3];
  const float* wq_b = (const float*)d_in[4];
  const float* wkv  = (const float*)d_in[5];
  const float* kvnw = (const float*)d_in[6];
  const float* wo_a = (const float*)d_in[7];
  const float* wo_b = (const float*)d_in[8];
  const float* sink = (const float*)d_in[9];
  float* out = (float*)d_out;
  char* ws = (char*)d_ws;

  // workspace layout (bytes), buffers reused once their producers/consumers retire
  bf16_t* xb     = (bf16_t*)(ws + 0);           //  8.0 MB  x bf16
  bf16_t* wqab   = (bf16_t*)(ws + 8388608);     //  6.0 MB
  bf16_t* wqbb   = (bf16_t*)(ws + 14680064);    //  9.0 MB
  bf16_t* wkvb   = (bf16_t*)(ws + 24117248);    //  0.75MB
  bf16_t* woab   = (bf16_t*)(ws + 24903680);    //  3.0 MB
  bf16_t* wobb   = (bf16_t*)(ws + 28049408);    //  8.0 MB
  float*  qloraf = (float*)(ws + 36438016);     // 12.0 MB (reused as obb)
  bf16_t* qlorab = (bf16_t*)(ws + 49020928);    //  6.0 MB
  float*  qf     = (float*)(ws + 55312384);     // 24.0 MB (reused as oabb)
  bf16_t* qbb    = (bf16_t*)(ws + 80478208);    // 12.0 MB
  float*  kvf    = (float*)(ws + 93061120);     //  1.5 MB
  bf16_t* kvbb   = (bf16_t*)(ws + 94633984);    //  0.75MB
  bf16_t* kvTb   = (bf16_t*)(ws + 95420416);    //  0.75MB -> total ~92 MB
  bf16_t* obb    = (bf16_t*)qloraf;             // attention output (qloraf dead)
  bf16_t* oabb   = (bf16_t*)qf;                 // oa bf16 (qf dead)

  auto cvt = [&](const float* src, bf16_t* dst, int n) {
    cvt_bf16<<<(n + 255) / 256, 256, 0, stream>>>(src, dst, n);
  };
  cvt(x, xb, T_TOK * HIDN);
  cvt(wq_a, wqab, QLRK * HIDN);
  cvt(wq_b, wqbb, NHEAD * HDIM * QLRK);
  cvt(wkv, wkvb, HDIM * HIDN);
  cvt(wo_a, woab, NGRP * OLRK * GDIM);
  cvt(wo_b, wobb, HIDN * NGRP * OLRK);

  // q_lora = x @ wq_a^T
  gemm_bf16<false><<<dim3(QLRK / 64, T_TOK / 64), 32, 0, stream>>>(
      xb, HIDN, wqab, HIDN, (void*)qloraf, QLRK, HIDN);
  rmsnorm_rows<<<T_TOK, 256, 0, stream>>>(qloraf, qnw, qlorab, QLRK);
  // q = q_lora @ wq_b^T
  gemm_bf16<false><<<dim3((NHEAD * HDIM) / 64, T_TOK / 64), 32, 0, stream>>>(
      qlorab, QLRK, wqbb, QLRK, (void*)qf, NHEAD * HDIM, QLRK);
  // kv = x @ wkv^T
  gemm_bf16<false><<<dim3(HDIM / 64, T_TOK / 64), 32, 0, stream>>>(
      xb, HIDN, wkvb, HIDN, (void*)kvf, HDIM, HIDN);
  q_norm_rope<<<T_TOK * NHEAD, 32, 0, stream>>>(qf, pos, qbb);
  kv_norm_rope<<<T_TOK, 32, 0, stream>>>(kvf, kvnw, pos, kvbb, kvTb);
  // causal MQA with sink
  attn_mqa<<<NHEAD * (T_TOK / 16), 32, 0, stream>>>(qbb, kvbb, kvTb, sink, obb);
  // grouped low-rank o projection: oa[:, g] = o_g @ wo_a[g]^T
  for (int g = 0; g < NGRP; ++g)
    gemm_bf16<true><<<dim3(OLRK / 64, T_TOK / 64), 32, 0, stream>>>(
        obb + g * GDIM, NHEAD * HDIM, woab + (size_t)g * OLRK * GDIM, GDIM,
        (void*)(oabb + g * OLRK), NGRP * OLRK, GDIM);
  // out = oa @ wo_b^T (f32)
  gemm_bf16<false><<<dim3(HIDN / 64, T_TOK / 64), 32, 0, stream>>>(
      oabb, NGRP * OLRK, wobb, NGRP * OLRK, (void*)out, HIDN, NGRP * OLRK);
}
